// HierAttnV3_42545946034821
// MI455X (gfx1250) — matrix operand order repaired and checked by reading, not compiled, not run on hardware
//
#include <hip/hip_runtime.h>

// ---------------------------------------------------------------------------
// Problem constants (match the reference)
// ---------------------------------------------------------------------------
static constexpr int Bc  = 4;
static constexpr int Tc  = 2048;
static constexpr int Ec  = 1024;
static constexpr int Hc  = 16;
static constexpr int DHc = 64;
static constexpr int Mc  = Bc * Tc;          // 8192 rows for all GEMMs

typedef __attribute__((ext_vector_type(16))) __bf16         v16bf;
typedef __attribute__((ext_vector_type(16))) unsigned short v16us;
typedef __attribute__((ext_vector_type(8)))  unsigned short v8us;
typedef __attribute__((ext_vector_type(8)))  float          v8f;

#define DEV __device__ __forceinline__

DEV unsigned short f2bf(float f) {
  unsigned int u = __builtin_bit_cast(unsigned int, f);
  u += 0x7FFFu + ((u >> 16) & 1u);          // round-to-nearest-even
  return (unsigned short)(u >> 16);
}
DEV float bf2f(unsigned short u) {
  unsigned int x = ((unsigned int)u) << 16;
  return __builtin_bit_cast(float, x);
}

// A fragment (16x32 bf16): lane = M row; lane-half hi holds K run
// [16*hi, 16*hi+16) (contiguous, matching B fragment).
DEV v16bf load_afrag_bf(const unsigned short* __restrict__ rowk0, int hi) {
  v8us lo = *(const v8us*)(rowk0 + 16 * hi);
  v8us hh = *(const v8us*)(rowk0 + 16 * hi + 8);
  v16us r;
#pragma unroll
  for (int i = 0; i < 8; ++i) { r[i] = lo[i]; r[8 + i] = hh[i]; }
  return __builtin_bit_cast(v16bf, r);
}

// B fragment (32x16 bf16): lane = N column, 16 contiguous K values.
DEV v16bf load_bfrag_bf(const unsigned short* __restrict__ p) {
  return __builtin_bit_cast(v16bf, *(const v16us*)p);
}

DEV v8f wmma_bf16(v16bf a, v16bf b, v8f c) {
  return __builtin_amdgcn_wmma_f32_16x16x32_bf16(false, a, false, b,
                                                 (short)0, c, false, false);
}

// ---------------------------------------------------------------------------
// Elementwise f32 -> bf16 conversion (x, weights, satt, ctx)
// ---------------------------------------------------------------------------
__global__ void conv_bf16_kernel(const float* __restrict__ in,
                                 unsigned short* __restrict__ out, int n) {
  int i = (blockIdx.x * blockDim.x + threadIdx.x) * 4;
  if (i >= n) return;
  float4 v = *(const float4*)(in + i);
  out[i + 0] = f2bf(v.x);
  out[i + 1] = f2bf(v.y);
  out[i + 2] = f2bf(v.z);
  out[i + 3] = f2bf(v.w);
}

// ---------------------------------------------------------------------------
// Transpose bf16 V (B,T,E) -> Vt (B,H,DH,T)
// ---------------------------------------------------------------------------
__global__ void vtrans_kernel(const unsigned short* __restrict__ V,
                              unsigned short* __restrict__ Vt) {
  int idx = blockIdx.x * blockDim.x + threadIdx.x;     // over B*H*DH*T
  if (idx >= Mc * Ec) return;
  int t   = idx & (Tc - 1);
  int row = idx / Tc;                                  // (b*H + h)*DH + dh
  int dh  = row & (DHc - 1);
  int hh  = (row / DHc) & (Hc - 1);
  int b   = row / (DHc * Hc);
  Vt[idx] = V[((size_t)(b * Tc + t)) * Ec + hh * DHc + dh];
}

// ---------------------------------------------------------------------------
// t = X @ Bm^T   (rank-8 LoRA inner product, M x 8; f32 inputs)
// ---------------------------------------------------------------------------
__global__ void lora_t_kernel(const float* __restrict__ X,
                              const float* __restrict__ Bm,
                              float* __restrict__ Tout, int K) {
  int idx = blockIdx.x * blockDim.x + threadIdx.x;
  if (idx >= Mc * 8) return;
  int rr = idx & 7;
  int m  = idx >> 3;
  const float* xr = X + (size_t)m * K;
  const float* br = Bm + (size_t)rr * K;
  float s = 0.0f;
  for (int k = 0; k < K; k += 4) {
    float4 xv = *(const float4*)(xr + k);
    float4 bv = *(const float4*)(br + k);
    s += xv.x * bv.x + xv.y * bv.y + xv.z * bv.z + xv.w * bv.w;
  }
  Tout[idx] = s;
}

// ---------------------------------------------------------------------------
// GEMM: Out = alpha * (X @ W^T [+ 2*(LT @ LA^T)]) , bf16 inputs, f32 accum.
// One wave computes a 16x64 strip (4 WMMA accumulators, A-frag reused 4x).
// Output: bf16 (OutB) or f32 (OutF, optional accumulate).
// ---------------------------------------------------------------------------
__global__ void __launch_bounds__(256)
gemm_kernel(const unsigned short* __restrict__ Xb,
            const unsigned short* __restrict__ Wb,
            const float* __restrict__ LT, const float* __restrict__ LA,
            float* __restrict__ OutF, unsigned short* __restrict__ OutB,
            const float* __restrict__ gatep,
            int M, int N, int K, int accumulate) {
  int ntn   = N >> 6;                         // 64-wide column strips
  int tile  = blockIdx.x * blockDim.y + threadIdx.y;
  int total = (M >> 4) * ntn;
  if (tile >= total) return;
  int tm = tile / ntn;
  int tn = tile % ntn;
  int lane = threadIdx.x;
  int r  = lane & 15;
  int hi = lane >> 4;

  const unsigned short* xr = Xb + (size_t)(tm * 16 + r) * K;
  const unsigned short* wr[4];
#pragma unroll
  for (int nt = 0; nt < 4; ++nt)
    wr[nt] = Wb + (size_t)(tn * 64 + nt * 16 + r) * K + 16 * hi;

  v8f acc[4];
#pragma unroll
  for (int nt = 0; nt < 4; ++nt) acc[nt] = (v8f){};

  for (int k = 0; k < K; k += 32) {
    v16bf a = load_afrag_bf(xr + k, hi);
#pragma unroll
    for (int nt = 0; nt < 4; ++nt) {
      v16bf bb = load_bfrag_bf(wr[nt] + k);
      acc[nt] = wmma_bf16(a, bb, acc[nt]);
    }
  }

  float Ar[4][8];
  if (LA != nullptr) {
#pragma unroll
    for (int nt = 0; nt < 4; ++nt) {
      const float* ap = LA + (size_t)(tn * 64 + nt * 16 + r) * 8;
#pragma unroll
      for (int i = 0; i < 8; ++i) Ar[nt][i] = ap[i];
    }
  }
  float alpha = 1.0f;
  if (gatep != nullptr) alpha = 1.0f / (1.0f + __expf(-gatep[0]));

#pragma unroll
  for (int j = 0; j < 8; ++j) {
    int m = tm * 16 + j + 8 * hi;
    float tp[8];
    if (LT != nullptr) {
      const float* tq = LT + (size_t)m * 8;
#pragma unroll
      for (int i = 0; i < 8; ++i) tp[i] = tq[i];
    }
#pragma unroll
    for (int nt = 0; nt < 4; ++nt) {
      float v = acc[nt][j];
      if (LT != nullptr) {
        float s = 0.0f;
#pragma unroll
        for (int i = 0; i < 8; ++i) s += tp[i] * Ar[nt][i];
        v += 2.0f * s;                        // LORA_S = 16/8
      }
      v *= alpha;
      size_t o = (size_t)m * N + tn * 64 + nt * 16 + r;
      if (OutB != nullptr) OutB[o] = f2bf(v);
      else if (accumulate)  OutF[o] += v;
      else                  OutF[o] = v;
    }
  }
}

// ---------------------------------------------------------------------------
// Flash-style attention, one wave per (b, h, 16-query tile).
// Q,K: bf16 (B,Tq/Tkv,E); Vt: bf16 (B,H,DH,Tkv). Out: f32 (B,Tq,E).
// ---------------------------------------------------------------------------
__global__ void __launch_bounds__(32)
attn_kernel(const unsigned short* __restrict__ Q,
            const unsigned short* __restrict__ Kp,
            const unsigned short* __restrict__ Vt,
            float* __restrict__ Outp, const float* __restrict__ slopes,
            int Tq, int Tkv, float scale, int causal, int accumulate) {
  const float FNEG = -3.0e38f;
  int nqt = Tq >> 4;
  int bid = blockIdx.x;
  int qt  = bid % nqt;
  int h   = (bid / nqt) % Hc;
  int b   = bid / (nqt * Hc);
  int lane = threadIdx.x;
  int r  = lane & 15;
  int hi = lane >> 4;
  int q0 = qt * 16;
  float slope = (slopes != nullptr) ? slopes[h] : 0.0f;

  __shared__ unsigned short Pl[16 * 32];     // P tile (16 q x 32 k) bf16

  const unsigned short* qp = Q + ((size_t)(b * Tq + q0 + r)) * Ec + h * DHc;
  v16bf qa0 = load_afrag_bf(qp, hi);
  v16bf qa1 = load_afrag_bf(qp + 32, hi);

  v8f oacc[4];
#pragma unroll
  for (int nt = 0; nt < 4; ++nt) oacc[nt] = (v8f){};
  float mstate[8], lstate[8];
#pragma unroll
  for (int j = 0; j < 8; ++j) { mstate[j] = FNEG; lstate[j] = 0.0f; }

  int nb = causal ? ((q0 + 16 + 31) >> 5) : ((Tkv + 31) >> 5);
  for (int kb = 0; kb < nb; ++kb) {
    int kbase = kb * 32;
    float sv[2][8];

#pragma unroll
    for (int st = 0; st < 2; ++st) {
      int key = kbase + st * 16 + r;
      const unsigned short* krow =
          Kp + ((size_t)(b * Tkv + key)) * Ec + h * DHc;
      v16bf b0 = load_bfrag_bf(krow + 16 * hi);
      v16bf b1 = load_bfrag_bf(krow + 32 + 16 * hi);
      v8f s = {};
      s = wmma_bf16(qa0, b0, s);
      s = wmma_bf16(qa1, b1, s);
#pragma unroll
      for (int j = 0; j < 8; ++j) {
        int qi = q0 + j + 8 * hi;
        float val = s[j] * scale + slope * (float)(qi - key);
        if (causal && key > qi) val = FNEG;
        sv[st][j] = val;
      }
    }

    // online softmax (rows live across 16-lane half-groups)
    float f[8];
#pragma unroll
    for (int j = 0; j < 8; ++j) {
      float rm = fmaxf(sv[0][j], sv[1][j]);
      rm = fmaxf(rm, __shfl_xor(rm, 1));
      rm = fmaxf(rm, __shfl_xor(rm, 2));
      rm = fmaxf(rm, __shfl_xor(rm, 4));
      rm = fmaxf(rm, __shfl_xor(rm, 8));
      float mnew = fmaxf(mstate[j], rm);
      f[j] = __expf(mstate[j] - mnew);
      mstate[j] = mnew;
      float p0 = __expf(sv[0][j] - mnew);
      float p1 = __expf(sv[1][j] - mnew);
      Pl[(j + 8 * hi) * 32 + r]      = f2bf(p0);
      Pl[(j + 8 * hi) * 32 + 16 + r] = f2bf(p1);
      float rs = p0 + p1;
      rs += __shfl_xor(rs, 1);
      rs += __shfl_xor(rs, 2);
      rs += __shfl_xor(rs, 4);
      rs += __shfl_xor(rs, 8);
      lstate[j] = lstate[j] * f[j] + rs;
    }
#pragma unroll
    for (int nt = 0; nt < 4; ++nt)
#pragma unroll
      for (int j = 0; j < 8; ++j) oacc[nt][j] *= f[j];

    __syncthreads();

    // Re-read P in A-frag layout
    v16bf pf;
    {
      const unsigned short* pr = Pl + r * 32;
      v16us t;
#pragma unroll
      for (int i = 0; i < 8; ++i) {
        t[i]     = pr[16 * hi + i];
        t[8 + i] = pr[16 * hi + 8 + i];
      }
      pf = __builtin_bit_cast(v16bf, t);
    }

    // O += P @ V  (B-frag = 16 contiguous keys from transposed V)
#pragma unroll
    for (int nt = 0; nt < 4; ++nt) {
      const unsigned short* vp =
          Vt + ((size_t)((b * Hc + h) * DHc + nt * 16 + r)) * Tkv +
          kbase + 16 * hi;
      oacc[nt] = wmma_bf16(pf, load_bfrag_bf(vp), oacc[nt]);
    }
    __syncthreads();
  }

#pragma unroll
  for (int nt = 0; nt < 4; ++nt)
#pragma unroll
    for (int j = 0; j < 8; ++j) {
      float val = oacc[nt][j] / lstate[j];
      size_t o = ((size_t)(b * Tq + q0 + j + 8 * hi)) * Ec + h * DHc +
                 nt * 16 + r;
      if (accumulate) Outp[o] += val; else Outp[o] = val;
    }
}

// ---------------------------------------------------------------------------
// pool1: 16-token chunk means (chunks 32..95) -> K1 bf16 (B,64,E),
//        V1t bf16 (B,H,DH,64)
// ---------------------------------------------------------------------------
__global__ void pool1_kernel(const unsigned short* __restrict__ Kf,
                             const unsigned short* __restrict__ Vf,
                             unsigned short* __restrict__ K1,
                             unsigned short* __restrict__ V1t) {
  int idx = blockIdx.x * blockDim.x + threadIdx.x;
  if (idx >= Bc * 64 * Ec) return;
  int e = idx & (Ec - 1);
  int i = (idx / Ec) & 63;
  int b = idx / (Ec * 64);
  int t0 = 512 + 16 * i;
  float sk = 0.0f, sv = 0.0f;
  for (int t = 0; t < 16; ++t) {
    size_t off = ((size_t)(b * Tc + t0 + t)) * Ec + e;
    sk += bf2f(Kf[off]);
    sv += bf2f(Vf[off]);
  }
  K1[idx] = f2bf(sk * (1.0f / 16.0f));
  int hh = e >> 6, dh = e & 63;
  V1t[((size_t)((b * Hc + hh) * DHc + dh)) * 64 + i] =
      f2bf(sv * (1.0f / 16.0f));
}

// ---------------------------------------------------------------------------
// pool2: 32-token means (== pair means of 16-token chunk means) ->
//        K2 bf16 (B,32,E), V2t bf16 (B,H,DH,32); also recon = 0.
// ---------------------------------------------------------------------------
__global__ void pool2_kernel(const unsigned short* __restrict__ Kf,
                             const unsigned short* __restrict__ Vf,
                             unsigned short* __restrict__ K2,
                             unsigned short* __restrict__ V2t,
                             float* __restrict__ recon) {
  int idx = blockIdx.x * blockDim.x + threadIdx.x;
  if (idx == 0) recon[0] = 0.0f;
  if (idx >= Bc * 32 * Ec) return;
  int e = idx & (Ec - 1);
  int j = (idx / Ec) & 31;
  int b = idx / (Ec * 32);
  int t0 = 512 + 32 * j;
  float sk = 0.0f, sv = 0.0f;
  for (int t = 0; t < 32; ++t) {
    size_t off = ((size_t)(b * Tc + t0 + t)) * Ec + e;
    sk += bf2f(Kf[off]);
    sv += bf2f(Vf[off]);
  }
  K2[idx] = f2bf(sk * (1.0f / 32.0f));
  int hh = e >> 6, dh = e & 63;
  V2t[((size_t)((b * Hc + hh) * DHc + dh)) * 32 + j] =
      f2bf(sv * (1.0f / 32.0f));
}

// ---------------------------------------------------------------------------
// Launcher
// ---------------------------------------------------------------------------
extern "C" void kernel_launch(void* const* d_in, const int* in_sizes, int n_in,
                              void* d_out, int out_size, void* d_ws,
                              size_t ws_size, hipStream_t stream) {
  (void)in_sizes; (void)n_in; (void)out_size; (void)ws_size;
  const float* x      = (const float*)d_in[0];
  const float* Wq_w   = (const float*)d_in[1];
  const float* Wq_A   = (const float*)d_in[2];
  const float* Wq_B   = (const float*)d_in[3];
  const float* Wk_w   = (const float*)d_in[4];
  const float* Wk_A   = (const float*)d_in[5];
  const float* Wk_B   = (const float*)d_in[6];
  const float* Wv_w   = (const float*)d_in[7];
  const float* Wv_A   = (const float*)d_in[8];
  const float* Wv_B   = (const float*)d_in[9];
  const float* proj_w = (const float*)d_in[10];
  const float* hq_w   = (const float*)d_in[11];
  const float* hp_w   = (const float*)d_in[12];
  const float* gate   = (const float*)d_in[13];
  const float* slopes = (const float*)d_in[14];
  float* out = (float*)d_out;

  const size_t SZ = (size_t)Mc * Ec;        // 8,388,608 elements
  const size_t WSZ = (size_t)Ec * Ec;       // 1,048,576 elements

  char* wsp = (char*)d_ws;
  auto alloc = [&](size_t bytes) {
    char* p = wsp;
    wsp += (bytes + 255) & ~(size_t)255;
    return p;
  };
  unsigned short* xb   = (unsigned short*)alloc(SZ * 2);
  unsigned short* wqb  = (unsigned short*)alloc(WSZ * 2);
  unsigned short* wkb  = (unsigned short*)alloc(WSZ * 2);
  unsigned short* wvb  = (unsigned short*)alloc(WSZ * 2);
  unsigned short* hqb  = (unsigned short*)alloc(WSZ * 2);
  unsigned short* prjb = (unsigned short*)alloc(WSZ * 2);
  unsigned short* hpb  = (unsigned short*)alloc(WSZ * 2);
  unsigned short* qb   = (unsigned short*)alloc(SZ * 2);
  unsigned short* kb   = (unsigned short*)alloc(SZ * 2);
  unsigned short* vb   = (unsigned short*)alloc(SZ * 2);
  unsigned short* qhb  = (unsigned short*)alloc(SZ * 2);
  unsigned short* vtb  = (unsigned short*)alloc(SZ * 2);
  float*          satt = (float*)alloc(SZ * 4);
  float*          ctx  = (float*)alloc(SZ * 4);
  unsigned short* sattb= (unsigned short*)alloc(SZ * 2);
  unsigned short* ctxb = (unsigned short*)alloc(SZ * 2);
  float*          tq   = (float*)alloc((size_t)Mc * 8 * 4);
  float*          tk   = (float*)alloc((size_t)Mc * 8 * 4);
  float*          tv   = (float*)alloc((size_t)Mc * 8 * 4);
  unsigned short* K1   = (unsigned short*)alloc((size_t)Bc * 64 * Ec * 2);
  unsigned short* V1t  = (unsigned short*)alloc((size_t)Bc * 64 * Ec * 2);
  unsigned short* K2   = (unsigned short*)alloc((size_t)Bc * 32 * Ec * 2);
  unsigned short* V2t  = (unsigned short*)alloc((size_t)Bc * 32 * Ec * 2);

  const float scale = 0.125f;               // 1/sqrt(64)
  dim3 gblk(32, 8);
  int gemm_grid = (Mc / 16) * (Ec / 64) / 8;  // 1024 blocks (16x64 strips)
  int attn_grid = Bc * Hc * (Tc / 16);        // 8192
  int cg_big    = (int)(SZ / 4 / 256);        // conv grid for M*E tensors
  int cg_w      = (int)(WSZ / 4 / 256);       // conv grid for E*E tensors

  // bf16 conversions of x and all weights
  conv_bf16_kernel<<<cg_big, 256, 0, stream>>>(x, xb, (int)SZ);
  conv_bf16_kernel<<<cg_w, 256, 0, stream>>>(Wq_w, wqb, (int)WSZ);
  conv_bf16_kernel<<<cg_w, 256, 0, stream>>>(Wk_w, wkb, (int)WSZ);
  conv_bf16_kernel<<<cg_w, 256, 0, stream>>>(Wv_w, wvb, (int)WSZ);
  conv_bf16_kernel<<<cg_w, 256, 0, stream>>>(hq_w, hqb, (int)WSZ);
  conv_bf16_kernel<<<cg_w, 256, 0, stream>>>(proj_w, prjb, (int)WSZ);
  conv_bf16_kernel<<<cg_w, 256, 0, stream>>>(hp_w, hpb, (int)WSZ);

  // rank-8 LoRA inner products (f32 path)
  lora_t_kernel<<<256, 256, 0, stream>>>(x, Wq_B, tq, Ec);
  lora_t_kernel<<<256, 256, 0, stream>>>(x, Wk_B, tk, Ec);
  lora_t_kernel<<<256, 256, 0, stream>>>(x, Wv_B, tv, Ec);

  // q/k/v (LoRA fused) and Qh projections -> bf16 outputs
  gemm_kernel<<<gemm_grid, gblk, 0, stream>>>(xb, wqb, tq, Wq_A, nullptr, qb,  nullptr, Mc, Ec, Ec, 0);
  gemm_kernel<<<gemm_grid, gblk, 0, stream>>>(xb, wkb, tk, Wk_A, nullptr, kb,  nullptr, Mc, Ec, Ec, 0);
  gemm_kernel<<<gemm_grid, gblk, 0, stream>>>(xb, wvb, tv, Wv_A, nullptr, vb,  nullptr, Mc, Ec, Ec, 0);
  gemm_kernel<<<gemm_grid, gblk, 0, stream>>>(xb, hqb, nullptr, nullptr, nullptr, qhb, nullptr, Mc, Ec, Ec, 0);

  // transpose V for contiguous P@V B-fragments
  vtrans_kernel<<<(int)(SZ / 256), 256, 0, stream>>>(vb, vtb);

  // standard causal ALiBi attention
  attn_kernel<<<attn_grid, 32, 0, stream>>>(qb, kb, vtb, satt, slopes,
                                            Tc, Tc, scale, 1, 0);

  // hierarchy pooling (+ recon scalar)
  pool1_kernel<<<(Bc * 64 * Ec) / 256, 256, 0, stream>>>(kb, vb, K1, V1t);
  pool2_kernel<<<(Bc * 32 * Ec) / 256, 256, 0, stream>>>(kb, vb, K2, V2t,
                                                         out + SZ);

  // hierarchical attends (sum of two softmax-attentions)
  attn_kernel<<<attn_grid, 32, 0, stream>>>(qhb, K1, V1t, ctx, nullptr,
                                            Tc, 64, scale, 0, 0);
  attn_kernel<<<attn_grid, 32, 0, stream>>>(qhb, K2, V2t, ctx, nullptr,
                                            Tc, 32, scale, 0, 1);

  // convert attention outputs for the final WMMA GEMMs
  conv_bf16_kernel<<<cg_big, 256, 0, stream>>>(satt, sattb, (int)SZ);
  conv_bf16_kernel<<<cg_big, 256, 0, stream>>>(ctx, ctxb, (int)SZ);

  // out = satt @ proj_w^T + sigmoid(gate) * (ctx @ hp_w^T)
  gemm_kernel<<<gemm_grid, gblk, 0, stream>>>(sattb, prjb, nullptr, nullptr, out, nullptr, nullptr, Mc, Ec, Ec, 0);
  gemm_kernel<<<gemm_grid, gblk, 0, stream>>>(ctxb,  hpb,  nullptr, nullptr, out, nullptr, gate,    Mc, Ec, Ec, 1);
}
